// ContextBERTSelfAttention_81638738362935
// MI455X (gfx1250) — matrix operand order repaired and checked by reading, not compiled, not run on hardware
//
#include <hip/hip_runtime.h>

// ---------------------------------------------------------------------------
// ContextBERT self-attention for MI455X (gfx1250, wave32, WMMA + TDM).
// B=8, S=1024, H=1024, NH=16, HD=64.
// bf16 WMMA GEMMs (fp32 accumulate) + flash attention (no score
// materialization) + Tensor Data Mover double-buffered K/V staging.
// ---------------------------------------------------------------------------

#define B_  8
#define S_  1024
#define H_  1024
#define NH_ 16
#define HD_ 64

typedef __attribute__((ext_vector_type(16))) __bf16 v16bf;
typedef __attribute__((ext_vector_type(8)))  __bf16 v8bf;
typedef __attribute__((ext_vector_type(8)))  float  v8f;

static __device__ __forceinline__ unsigned short f2bf_bits(float f) {
  unsigned u = __float_as_uint(f);
  return (unsigned short)((u + 0x7FFFu + ((u >> 16) & 1u)) >> 16);  // RNE
}
static __device__ __forceinline__ float bf2f(unsigned short b) {
  return __uint_as_float(((unsigned)b) << 16);
}
static __device__ __forceinline__ v8f zero8() {
  v8f z = {0.f, 0.f, 0.f, 0.f, 0.f, 0.f, 0.f, 0.f};
  return z;
}
static __device__ __forceinline__ v16bf cat16(v8bf lo, v8bf hi) {
  return __builtin_shufflevector(lo, hi, 0, 1, 2, 3, 4, 5, 6, 7,
                                 8, 9, 10, 11, 12, 13, 14, 15);
}
static __device__ __forceinline__ v8f wmma_bf16(v16bf a, v16bf b, v8f c) {
  return __builtin_amdgcn_wmma_f32_16x16x32_bf16(false, a, false, b,
                                                 (short)0, c, false, false);
}

// ---------------------------------------------------------------------------
// Tensor Data Mover: async 2D tile (bf16 elements) global -> LDS.
// D# packing per CDNA5 ISA ch.8 (group0: count/lds/global/type,
// group1: data_size, tensor dims, tile dims, dim0 stride).
// This toolchain exposes the 6-arg builtin form:
//   (u32x4 g0, i32x8 g1, i32x4 g2, i32x4 g3, i32x8 extra, i32 cpol)
// ---------------------------------------------------------------------------
#if __has_builtin(__builtin_amdgcn_tensor_load_to_lds)
#define HAVE_TDM 1
typedef __attribute__((ext_vector_type(4))) unsigned int u32x4;
typedef __attribute__((ext_vector_type(8))) int i32x8;
typedef __attribute__((ext_vector_type(4))) int i32x4;

static __device__ __forceinline__ void tdm_load_2d_bf16(
    unsigned lds_addr, unsigned long long gaddr, unsigned tile_d0,
    unsigned tile_d1, unsigned long long stride0_elems) {
  u32x4 g0;
  g0[0] = 1u;  // count=1 valid descriptor, gather off
  g0[1] = lds_addr;
  g0[2] = (unsigned)gaddr;
  g0[3] = (unsigned)((gaddr >> 32) & 0x01FFFFFFu) | 0x80000000u;  // type=2
  const unsigned td0 = 0x40000000u, td1 = 0x40000000u;  // in-bounds always
  i32x8 g1;
  g1[0] = (int)(1u << 16);  // data_size = 2 bytes (bf16)
  g1[1] = (int)((td0 & 0xFFFFu) << 16);
  g1[2] = (int)((td0 >> 16) | ((td1 & 0xFFFFu) << 16));
  g1[3] = (int)((td1 >> 16) | (tile_d0 << 16));
  g1[4] = (int)(tile_d1 & 0xFFFFu);  // tile_dim1 | tile_dim2(=0)<<16
  g1[5] = (int)(unsigned)(stride0_elems & 0xFFFFFFFFull);
  g1[6] = (int)(unsigned)((stride0_elems >> 32) & 0xFFFFull);
  g1[7] = 0;
  i32x4 zz4;
  zz4[0] = zz4[1] = zz4[2] = zz4[3] = 0;
  i32x8 zz8;
#pragma unroll
  for (int i = 0; i < 8; ++i) zz8[i] = 0;
  __builtin_amdgcn_tensor_load_to_lds(g0, g1, zz4, zz4, zz8, 0);
}

static __device__ __forceinline__ void tdm_wait0() {
#if __has_builtin(__builtin_amdgcn_s_wait_tensorcnt)
  __builtin_amdgcn_s_wait_tensorcnt(0);
#else
  asm volatile("s_wait_tensorcnt 0x0" ::: "memory");
#endif
}
#else
#define HAVE_TDM 0
#endif

// ---------------------------------------------------------------------------
// Kernel 1: X[8192x1024] @ W[1024x1024] + bias -> bf16.
// z in {0,1,2} -> (Wq,bq,q [B,NH,S,HD]) / (Wk,bk,k [B,NH,S,HD]) /
//                 (Wv,bv,v [B,NH,HD,S] d-major, so flash can TDM it directly).
// Block tile 128x128, K-step 32, 8 waves each owning a 64x32 sub-tile.
// ---------------------------------------------------------------------------
__global__ __launch_bounds__(256) void proj_gemm(
    const float* __restrict__ X,
    const float* __restrict__ Wq, const float* __restrict__ Wk,
    const float* __restrict__ Wv,
    const float* __restrict__ bq, const float* __restrict__ bk,
    const float* __restrict__ bv,
    unsigned short* __restrict__ q_bf, unsigned short* __restrict__ k_bf,
    unsigned short* __restrict__ v_bf) {
  const int z = blockIdx.z;
  const float* W = (z == 0) ? Wq : ((z == 1) ? Wk : Wv);
  const float* bias = (z == 0) ? bq : ((z == 1) ? bk : bv);
  unsigned short* dst = (z == 0) ? q_bf : ((z == 1) ? k_bf : v_bf);

  const int n0 = blockIdx.x * 128;
  const int m0 = blockIdx.y * 128;
  const int tid = threadIdx.x;
  const int wave = tid >> 5;
  const int lane = tid & 31;
  const int half = lane >> 4;
  const int l16 = lane & 15;
  const int wm = wave >> 2;  // 0..1  (64 rows each)
  const int wn = wave & 3;   // 0..3  (32 cols each)

  __shared__ unsigned short sA[128 * 40];   // A tile, row-major [m][k]
  __shared__ unsigned short sBt[128 * 40];  // B tile, TRANSPOSED [n][k]

  v8f acc[4][2];
#pragma unroll
  for (int i = 0; i < 4; ++i)
#pragma unroll
    for (int j = 0; j < 2; ++j) acc[i][j] = zero8();

  for (int k0 = 0; k0 < H_; k0 += 32) {
    {
      int row = tid >> 1;
      int kp = (tid & 1) << 4;
      const float4* s4 = (const float4*)(X + (size_t)(m0 + row) * H_ + k0 + kp);
      float4 f0 = s4[0], f1 = s4[1], f2 = s4[2], f3 = s4[3];
      float vals[16] = {f0.x, f0.y, f0.z, f0.w, f1.x, f1.y, f1.z, f1.w,
                        f2.x, f2.y, f2.z, f2.w, f3.x, f3.y, f3.z, f3.w};
      unsigned* dp = (unsigned*)&sA[row * 40 + kp];
#pragma unroll
      for (int i = 0; i < 8; ++i) {
        unsigned lo = f2bf_bits(vals[2 * i]);
        unsigned hi = f2bf_bits(vals[2 * i + 1]);
        dp[i] = lo | (hi << 16);
      }
    }
    {
      int kk = tid >> 3;        // 0..31
      int np = (tid & 7) << 4;  // 0..112
      const float4* s4 = (const float4*)(W + (size_t)(k0 + kk) * H_ + n0 + np);
      float4 f0 = s4[0], f1 = s4[1], f2 = s4[2], f3 = s4[3];
      float vals[16] = {f0.x, f0.y, f0.z, f0.w, f1.x, f1.y, f1.z, f1.w,
                        f2.x, f2.y, f2.z, f2.w, f3.x, f3.y, f3.z, f3.w};
#pragma unroll
      for (int i = 0; i < 16; ++i)
        sBt[(np + i) * 40 + kk] = f2bf_bits(vals[i]);
    }
    __syncthreads();

    v16bf bfrag[2];
#pragma unroll
    for (int nf = 0; nf < 2; ++nf) {
      const unsigned short* brow = &sBt[(wn * 32 + nf * 16 + l16) * 40];
      bfrag[nf] = cat16(*(const v8bf*)(brow + half * 8),
                        *(const v8bf*)(brow + 16 + half * 8));
    }
#pragma unroll
    for (int mf = 0; mf < 4; ++mf) {
      const unsigned short* arow = &sA[(wm * 64 + mf * 16 + l16) * 40];
      v16bf afrag = cat16(*(const v8bf*)(arow + half * 8),
                          *(const v8bf*)(arow + 16 + half * 8));
      acc[mf][0] = wmma_bf16(afrag, bfrag[0], acc[mf][0]);
      acc[mf][1] = wmma_bf16(afrag, bfrag[1], acc[mf][1]);
    }
    __syncthreads();
  }

#pragma unroll
  for (int mf = 0; mf < 4; ++mf)
#pragma unroll
    for (int nf = 0; nf < 2; ++nf)
#pragma unroll
      for (int v = 0; v < 8; ++v) {
        int grow = m0 + wm * 64 + mf * 16 + half * 8 + v;  // b*S+s
        int gcol = n0 + wn * 32 + nf * 16 + l16;           // h*HD+d
        float val = acc[mf][nf][v] + bias[gcol];
        int bb = grow >> 10, ss = grow & 1023;
        int hh = gcol >> 6, dd = gcol & 63;
        size_t idx = (z == 2)
                         ? ((((size_t)bb * NH_ + hh) * HD_ + dd) * S_ + ss)
                         : ((((size_t)bb * NH_ + hh) * S_ + ss) * HD_ + dd);
        dst[idx] = f2bf_bits(val);
      }
}

// ---------------------------------------------------------------------------
// Kernel 2: context gating (fp32 VALU 64x64 GEMM + sigmoid). grid.z: 0=q, 1=k.
// ---------------------------------------------------------------------------
__global__ __launch_bounds__(256) void gating(
    const float* __restrict__ context,
    const unsigned short* __restrict__ q_bf,
    const unsigned short* __restrict__ k_bf,
    const float* __restrict__ Wcq, const float* __restrict__ bcq,
    const float* __restrict__ Wck, const float* __restrict__ bck,
    const float* __restrict__ w_lqc, const float* __restrict__ w_lqq,
    const float* __restrict__ w_lkc, const float* __restrict__ w_lkk,
    unsigned short* __restrict__ qc_bf, unsigned short* __restrict__ kc_bf) {
  const int z = blockIdx.z;
  const float* W = (z == 0) ? Wcq : Wck;
  const float* bi = (z == 0) ? bcq : bck;
  const float* wc = (z == 0) ? w_lqc : w_lkc;
  const float* wg = (z == 0) ? w_lqq : w_lkk;
  const unsigned short* src = (z == 0) ? q_bf : k_bf;
  unsigned short* dst = (z == 0) ? qc_bf : kc_bf;

  const int bh = blockIdx.x;
  const int b = bh >> 4, h = bh & 15;
  const int tid = threadIdx.x;
  const int ty = tid >> 6;
  const int tx = tid & 63;

  __shared__ float sW[64 * 64];
  __shared__ float sb[64], swc[64], swg[64];
  __shared__ float sce[4][64];
  __shared__ float ssum[4][2];

  for (int i = tid; i < 64 * 64; i += 256) sW[i] = W[i];
  if (tid < 64) {
    sb[tid] = bi[tid];
    swc[tid] = wc[tid];
    swg[tid] = wg[tid];
  }
  __syncthreads();

  const int rowBase = blockIdx.y * 64;
  for (int it = 0; it < 16; ++it) {
    int s = rowBase + it * 4 + ty;
    float cev = context[((size_t)b * S_ + s) * H_ + h * 64 + tx];
    sce[ty][tx] = cev;
    __syncthreads();

    float acc = sb[tx];
#pragma unroll 8
    for (int j = 0; j < 64; ++j) acc = fmaf(sce[ty][j], sW[j * 64 + tx], acc);

    size_t qidx = ((size_t)bh * S_ + s) * HD_ + tx;
    float qv = bf2f(src[qidx]);

    float part = acc * swc[tx] + qv * swg[tx];
#pragma unroll
    for (int o = 16; o >= 1; o >>= 1) part += __shfl_xor(part, o);
    if ((tx & 31) == 0) ssum[ty][tx >> 5] = part;
    __syncthreads();
    float lam = 1.f / (1.f + __expf(-(ssum[ty][0] + ssum[ty][1])));

    float outv = (1.f - lam) * qv + lam * acc;
    dst[qidx] = f2bf_bits(outv);
    __syncthreads();
  }
}

// ---------------------------------------------------------------------------
// Kernel 3: flash attention per (b,h). 128 query rows/block, 8 waves x 16
// rows. K block [32 keys][64 d] and V block [64 d][32 keys] TDM-DMA'd into
// double-buffered LDS (async, overlapped one full iteration ahead).
// Online softmax in registers; P transposed C->A via per-wave LDS with
// s_wait_dscnt. v is stored d-major [B,NH,HD,S] so its TDM tile lands in
// exactly the B-fragment layout for the PV WMMA.
// ---------------------------------------------------------------------------
__global__ __launch_bounds__(256) void flash_attn(
    const unsigned short* __restrict__ qc,
    const unsigned short* __restrict__ kc,
    const unsigned short* __restrict__ vb,
    const float* __restrict__ mask,
    float* __restrict__ out) {
  const int qblk = blockIdx.x;  // 0..7
  const int bh = blockIdx.y;    // 0..127
  const int b = bh >> 4, h = bh & 15;
  const int tid = threadIdx.x;
  const int wave = tid >> 5;
  const int lane = tid & 31;
  const int half = lane >> 4;
  const int l16 = lane & 15;

  __shared__ unsigned short sK[2][32 * 64];   // [key][d]
  __shared__ unsigned short sVt[2][64 * 32];  // [d][key]
  __shared__ unsigned short sP[8][16 * 32];   // per-wave P transpose buffer

  const int rowBase = qblk * 128 + wave * 16;

  const unsigned short* qrow = qc + ((size_t)bh * S_ + rowBase + l16) * HD_;
  v16bf Qa0 = cat16(*(const v8bf*)(qrow + half * 8),
                    *(const v8bf*)(qrow + 16 + half * 8));
  v16bf Qa1 = cat16(*(const v8bf*)(qrow + 32 + half * 8),
                    *(const v8bf*)(qrow + 48 + half * 8));

  v8f O[4];
#pragma unroll
  for (int f = 0; f < 4; ++f) O[f] = zero8();
  float mstat[8], lstat[8];
#pragma unroll
  for (int v = 0; v < 8; ++v) {
    mstat[v] = -1e30f;
    lstat[v] = 0.f;
  }

  const float scale = 0.125f;  // 1/sqrt(64)
  const unsigned short* kbase = kc + (size_t)bh * S_ * HD_;   // [S][HD]
  const unsigned short* vbase = vb + (size_t)bh * HD_ * S_;   // [HD][S]
  const float* mrow = mask + (size_t)b * S_;

#if HAVE_TDM
  const unsigned sK_lds = (unsigned)(size_t)(&sK[0][0]);
  const unsigned sVt_lds = (unsigned)(size_t)(&sVt[0][0]);
  if (wave == 0) {
    // prologue: DMA first K/V blocks into buffer 0
    tdm_load_2d_bf16(sK_lds, (unsigned long long)(size_t)kbase,
                     /*tile_d0=*/2048, /*tile_d1=*/1, /*stride=*/2048);
    tdm_load_2d_bf16(sVt_lds, (unsigned long long)(size_t)vbase,
                     /*tile_d0=*/32, /*tile_d1=*/64, /*stride=*/S_);
  }
#endif

  for (int j = 0; j < S_; j += 32) {
    const int buf = (j >> 5) & 1;
#if HAVE_TDM
    if (wave == 0) tdm_wait0();  // current block landed
    __syncthreads();             // everyone sees it; prev buffer now free
    if (wave == 0 && (j + 32) < S_) {
      // overlap next block's DMA under this iteration's compute
      tdm_load_2d_bf16(
          sK_lds + (unsigned)((buf ^ 1) * 32 * 64 * 2),
          (unsigned long long)(size_t)(kbase + (size_t)(j + 32) * HD_),
          2048, 1, 2048);
      tdm_load_2d_bf16(
          sVt_lds + (unsigned)((buf ^ 1) * 64 * 32 * 2),
          (unsigned long long)(size_t)(vbase + (size_t)(j + 32)),
          32, 64, S_);
    }
#else
    ((uint4*)sK[buf])[tid] = ((const uint4*)(kbase + (size_t)j * HD_))[tid];
    {
      int e = tid * 8;        // 64 rows (d) x 32 keys, 8 contiguous each
      int r = e >> 5;
      int c0 = e & 31;
      *(uint4*)&sVt[buf][e] =
          *(const uint4*)(vbase + (size_t)r * S_ + j + c0);
    }
    __syncthreads();
#endif

    // --- scores: two 16x16 key tiles ---
    v8f S0 = zero8(), S1 = zero8();
    {
      const unsigned short* kr0 = &sK[buf][(0 * 16 + l16) * 64];
      const unsigned short* kr1 = &sK[buf][(1 * 16 + l16) * 64];
      v16bf Bk;
      Bk = cat16(*(const v8bf*)(kr0 + half * 8),
                 *(const v8bf*)(kr0 + 16 + half * 8));
      S0 = wmma_bf16(Qa0, Bk, S0);
      Bk = cat16(*(const v8bf*)(kr0 + 32 + half * 8),
                 *(const v8bf*)(kr0 + 48 + half * 8));
      S0 = wmma_bf16(Qa1, Bk, S0);
      Bk = cat16(*(const v8bf*)(kr1 + half * 8),
                 *(const v8bf*)(kr1 + 16 + half * 8));
      S1 = wmma_bf16(Qa0, Bk, S1);
      Bk = cat16(*(const v8bf*)(kr1 + 32 + half * 8),
                 *(const v8bf*)(kr1 + 48 + half * 8));
      S1 = wmma_bf16(Qa1, Bk, S1);
    }

    float mk0 = mrow[j + l16];
    float mk1 = mrow[j + 16 + l16];
    float s0v[8], s1v[8], rmax[8], rs[8];
#pragma unroll
    for (int v = 0; v < 8; ++v) {
      s0v[v] = S0[v] * scale + mk0;
      s1v[v] = S1[v] * scale + mk1;
      rmax[v] = fmaxf(s0v[v], s1v[v]);
    }
#pragma unroll
    for (int o = 8; o >= 1; o >>= 1)
#pragma unroll
      for (int v = 0; v < 8; ++v)
        rmax[v] = fmaxf(rmax[v], __shfl_xor(rmax[v], o));

#pragma unroll
    for (int v = 0; v < 8; ++v) {
      float mn = fmaxf(mstat[v], rmax[v]);
      float alpha = __expf(mstat[v] - mn);
      mstat[v] = mn;
      s0v[v] = __expf(s0v[v] - mn);
      s1v[v] = __expf(s1v[v] - mn);
      rs[v] = s0v[v] + s1v[v];
      lstat[v] *= alpha;
#pragma unroll
      for (int f = 0; f < 4; ++f) O[f][v] *= alpha;
    }
#pragma unroll
    for (int o = 8; o >= 1; o >>= 1)
#pragma unroll
      for (int v = 0; v < 8; ++v) rs[v] += __shfl_xor(rs[v], o);
#pragma unroll
    for (int v = 0; v < 8; ++v) lstat[v] += rs[v];

    // --- P (C-layout) -> per-wave LDS -> A-layout bf16 fragment ---
    unsigned short* pw = &sP[wave][0];
#pragma unroll
    for (int v = 0; v < 8; ++v) {
      pw[(v + half * 8) * 32 + l16] = f2bf_bits(s0v[v]);
      pw[(v + half * 8) * 32 + 16 + l16] = f2bf_bits(s1v[v]);
    }
    asm volatile("s_wait_dscnt 0x0" ::: "memory");
    v16bf Pa = cat16(*(const v8bf*)(pw + l16 * 32 + half * 8),
                     *(const v8bf*)(pw + l16 * 32 + 16 + half * 8));

    // --- O += P(16x32) @ V(32x64) ---
#pragma unroll
    for (int f = 0; f < 4; ++f) {
      const unsigned short* vr = &sVt[buf][(f * 16 + l16) * 32];
      v16bf Bv = cat16(*(const v8bf*)(vr + half * 8),
                       *(const v8bf*)(vr + 16 + half * 8));
      O[f] = wmma_bf16(Pa, Bv, O[f]);
    }
#if !HAVE_TDM
    __syncthreads();
#endif
  }

  // --- epilogue: normalize and write [B,S,H] fp32 ---
#pragma unroll
  for (int f = 0; f < 4; ++f)
#pragma unroll
    for (int v = 0; v < 8; ++v) {
      int srow = rowBase + v + half * 8;
      int col = h * 64 + f * 16 + l16;
      out[((size_t)b * S_ + srow) * H_ + col] = O[f][v] / lstat[v];
    }
}

// ---------------------------------------------------------------------------
extern "C" void kernel_launch(void* const* d_in, const int* in_sizes, int n_in,
                              void* d_out, int out_size, void* d_ws,
                              size_t ws_size, hipStream_t stream) {
  (void)in_sizes; (void)n_in; (void)out_size; (void)ws_size;
  const float* hidden = (const float*)d_in[0];
  const float* mask = (const float*)d_in[1];
  const float* ctx = (const float*)d_in[2];
  const float* Wq = (const float*)d_in[3];
  const float* bq = (const float*)d_in[4];
  const float* Wk = (const float*)d_in[5];
  const float* bk = (const float*)d_in[6];
  const float* Wv = (const float*)d_in[7];
  const float* bv = (const float*)d_in[8];
  const float* Wcq = (const float*)d_in[9];
  const float* bcq = (const float*)d_in[10];
  const float* Wck = (const float*)d_in[11];
  const float* bck = (const float*)d_in[12];
  const float* w_lqc = (const float*)d_in[13];
  const float* w_lqq = (const float*)d_in[14];
  const float* w_lkc = (const float*)d_in[15];
  const float* w_lkk = (const float*)d_in[16];
  float* out = (float*)d_out;

  const size_t elems = (size_t)B_ * NH_ * S_ * HD_;
  unsigned short* q_bf = (unsigned short*)d_ws;
  unsigned short* k_bf = q_bf + elems;
  unsigned short* v_bf = k_bf + elems;   // [B,NH,HD,S] d-major
  unsigned short* qc_bf = v_bf + elems;
  unsigned short* kc_bf = qc_bf + elems;

  proj_gemm<<<dim3(H_ / 128, (B_ * S_) / 128, 3), 256, 0, stream>>>(
      hidden, Wq, Wk, Wv, bq, bk, bv, q_bf, k_bf, v_bf);
  gating<<<dim3(B_ * NH_, S_ / 64, 2), 256, 0, stream>>>(
      ctx, q_bf, k_bf, Wcq, bcq, Wck, bck, w_lqc, w_lqq, w_lkc, w_lkk, qc_bf,
      kc_bf);
  flash_attn<<<dim3(S_ / 128, B_ * NH_), 256, 0, stream>>>(qc_bf, kc_bf, v_bf,
                                                           mask, out);
}
